// _MoEMLP_2027224563966
// MI455X (gfx1250) — compile-verified, matrix operand order checked by reading, not run on hardware
//
#include <hip/hip_runtime.h>
#include <hip/hip_bf16.h>

// Problem dims (fixed by the reference)
#define DN 1024   // tokens (B*S)
#define DH 1024   // hidden
#define DI 1024   // intermediate
#define DE 8      // experts

// LDS tile geometry: 64 rows x 64 K, padded row stride 72 bf16 (144 B).
// 144 is a multiple of 16 -> aligned ds_load_b128; 36*r mod 64 is a
// permutation over r=0..15 -> conflict-free per-lane fragment rows.
#define KTILE   64
#define TSTRIDE 72
#define TILE_ELEMS (64 * TSTRIDE)

typedef __bf16 bf16;
typedef __attribute__((ext_vector_type(16))) bf16  v16bf;
typedef __attribute__((ext_vector_type(8)))  bf16  v8bf;
typedef __attribute__((ext_vector_type(8)))  float v8f;

__device__ __forceinline__ unsigned pack_bf16(float a, float b) {
    union { unsigned u; bf16 h[2]; } p;
    p.h[0] = (bf16)a;
    p.h[1] = (bf16)b;
    return p.u;
}

// Fragment load for V_WMMA_*_16X16X32_BF16 from an LDS tile laid out
// row-major [rows][TSTRIDE].  Per ISA: lanes 0-15 hold K {0..7,16..23},
// lanes 16-31 hold K {8..15,24..31}.  Two aligned ds_load_b128 each.
__device__ __forceinline__ v16bf load_frag(const bf16* tile, int row, int ks,
                                           int lane) {
    const bf16* p = tile + row * TSTRIDE + ks;
    const int k0 = (lane & 16) ? 8 : 0;
    union { v16bf v; v8bf h[2]; } u;
    u.h[0] = *(const v8bf*)(p + k0);
    u.h[1] = *(const v8bf*)(p + k0 + 16);
    return u.v;
}

__device__ __forceinline__ v8f wmma_bf16(v16bf a, v16bf b, v8f c) {
    return __builtin_amdgcn_wmma_f32_16x16x32_bf16(
        false, a, false, b, (short)0, c, false, false);
}

// Async 16-byte memory->LDS copy (gfx1250, ASYNCcnt-tracked).
__device__ __forceinline__ void async_copy_b128(unsigned lds_byte_off,
                                                const void* gaddr) {
    asm volatile("global_load_async_to_lds_b128 %0, %1, off"
                 :: "v"(lds_byte_off), "v"((unsigned long long)gaddr)
                 : "memory");
}
__device__ __forceinline__ void wait_async0() {
    asm volatile("s_wait_asynccnt 0" ::: "memory");
}
__device__ __forceinline__ unsigned lds_off(const void* p) {
    return (unsigned)(unsigned long long)p;
}

// ---------------------------------------------------------------------------
// Kernel 0: hidden_states fp32 -> bf16 (done once; gate_up reuses it via
// async DMA instead of reconverting the same tile 128x).
// ---------------------------------------------------------------------------
__global__ __launch_bounds__(256) void moe_cvt_k(
    const float* __restrict__ x, unsigned* __restrict__ xh /* bf16 pairs */) {
    const size_t t = (size_t)blockIdx.x * 256 + threadIdx.x;   // 8 floats each
    const float2* src = (const float2*)(x + t * 8);
    uint4 d;
    d.x = pack_bf16(src[0].x, src[0].y);
    d.y = pack_bf16(src[1].x, src[1].y);
    d.z = pack_bf16(src[2].x, src[2].y);
    d.w = pack_bf16(src[3].x, src[3].y);
    *(uint4*)(xh + t * 4) = d;
}

// ---------------------------------------------------------------------------
// Kernel 1: router.  logits = x @ rw^T, softmax, top-2.
// ---------------------------------------------------------------------------
__global__ __launch_bounds__(256) void moe_router_k(
    const float* __restrict__ x,        // [N,H]
    const float* __restrict__ rw,       // [E,H]
    float* __restrict__ scores_out,     // [N,E]
    float* __restrict__ wmask)          // [N,E]
{
    const int tid = threadIdx.x;
    const int t   = blockIdx.x * 32 + (tid >> 3);
    const int e   = tid & 7;

    const float* xr = x  + (size_t)t * DH;
    const float* wr = rw + (size_t)e * DH;
    float acc = 0.f;
    #pragma unroll 4
    for (int h = 0; h < DH; ++h) acc = fmaf(xr[h], wr[h], acc);

    __shared__ float lg[32][8];
    lg[tid >> 3][e] = acc;
    __syncthreads();

    if (e == 0) {
        float* row = lg[tid >> 3];
        float mx = row[0];
        #pragma unroll
        for (int j = 1; j < DE; ++j) mx = fmaxf(mx, row[j]);
        float sc[DE]; float s = 0.f;
        #pragma unroll
        for (int j = 0; j < DE; ++j) { sc[j] = __expf(row[j] - mx); s += sc[j]; }
        const float inv = 1.f / s;
        int   b0 = 0, b1 = -1;
        float v0 = -1.f, v1 = -1.f;
        #pragma unroll
        for (int j = 0; j < DE; ++j) {
            sc[j] *= inv;
            scores_out[(size_t)t * DE + j] = sc[j];
            if (sc[j] > v0)      { v1 = v0; b1 = b0; v0 = sc[j]; b0 = j; }
            else if (sc[j] > v1) { v1 = sc[j]; b1 = j; }
        }
        #pragma unroll
        for (int j = 0; j < DE; ++j)
            wmask[(size_t)t * DE + j] = (j == b0) ? v0 : ((j == b1) ? v1 : 0.f);
    }
}

// ---------------------------------------------------------------------------
// Kernel 2: gate_up GEMM + clamp/GLU activation, pre-scaled by router weight.
// Double-buffered: A panels async-DMA'd from bf16 x; B panels register-staged
// (issue loads -> WMMA on current buffers -> convert+store), 1 barrier/iter.
// ---------------------------------------------------------------------------
__global__ __launch_bounds__(256) void moe_gateup_k(
    const bf16*  __restrict__ xh,       // [N,H] bf16
    const float* __restrict__ gup,      // [E,H,2I]
    const float* __restrict__ gub,      // [E,2I]
    const float* __restrict__ wmask,    // [N,E]
    bf16* __restrict__ actw)            // [N,E,I]
{
    const int n0 = blockIdx.x * 64;
    const int i0 = blockIdx.y * 64;
    const int e  = blockIdx.z;

    __shared__ bf16 As[2][TILE_ELEMS];
    __shared__ bf16 Bg[2][TILE_ELEMS];
    __shared__ bf16 Bu[2][TILE_ELEMS];

    const int tid   = threadIdx.x;
    const int lane  = tid & 31;
    const int wave  = tid >> 5;
    const int rBase = (wave >> 1) * 16;   // 4 row-waves
    const int cBase = (wave & 1) * 32;    // 2 col-waves
    const int m     = lane & 15;
    const int hi    = (lane >> 4) & 1;

    v8f accG0 = {}, accG1 = {}, accU0 = {}, accU1 = {};
    float2 t0[8], t1[8];

    // --- staging helpers (fully unrolled; arrays live in VGPRs) ---
    auto issueA = [&](int buf, int k0) {
        const unsigned base = lds_off(&As[buf][0]);
        #pragma unroll
        for (int it = 0; it < 2; ++it) {
            const int idx = tid + it * 256;           // 512 x 16B chunks
            const int r = idx >> 3, q = idx & 7;
            async_copy_b128(base + (unsigned)(r * (TSTRIDE * 2) + q * 16),
                            xh + (size_t)(n0 + r) * DH + k0 + q * 8);
        }
    };
    auto issueB = [&](int k0) {
        #pragma unroll
        for (int it = 0; it < 8; ++it) {
            const int idx = tid + it * 256;           // 2048 uints per buffer
            const int kp = idx >> 6, cc = idx & 63;   // k = kp*2
            const size_t base =
                ((size_t)e * DH + (k0 + kp * 2)) * (2 * DI) + 2 * (i0 + cc);
            t0[it] = *(const float2*)&gup[base];           // row k
            t1[it] = *(const float2*)&gup[base + 2 * DI];  // row k+1
        }
    };
    auto storeB = [&](int buf) {
        unsigned* g = (unsigned*)&Bg[buf][0];
        unsigned* u = (unsigned*)&Bu[buf][0];
        #pragma unroll
        for (int it = 0; it < 8; ++it) {
            const int idx = tid + it * 256;
            const int kp = idx >> 6, cc = idx & 63;
            g[cc * (TSTRIDE / 2) + kp] = pack_bf16(t0[it].x, t1[it].x);
            u[cc * (TSTRIDE / 2) + kp] = pack_bf16(t0[it].y, t1[it].y);
        }
    };
    auto compute = [&](int buf) {
        #pragma unroll
        for (int ks = 0; ks < KTILE; ks += 32) {
            const v16bf a  = load_frag(&As[buf][0], rBase + m,      ks, lane);
            const v16bf g0 = load_frag(&Bg[buf][0], cBase + m,      ks, lane);
            const v16bf g1 = load_frag(&Bg[buf][0], cBase + 16 + m, ks, lane);
            const v16bf u0 = load_frag(&Bu[buf][0], cBase + m,      ks, lane);
            const v16bf u1 = load_frag(&Bu[buf][0], cBase + 16 + m, ks, lane);
            accG0 = wmma_bf16(a, g0, accG0);
            accG1 = wmma_bf16(a, g1, accG1);
            accU0 = wmma_bf16(a, u0, accU0);
            accU1 = wmma_bf16(a, u1, accU1);
        }
    };

    // Prologue: fill buffer 0.
    issueA(0, 0);
    issueB(0);
    storeB(0);
    wait_async0();
    __syncthreads();

    const int NT = DH / KTILE;            // 16
    for (int kt = 0; kt < NT; ++kt) {
        const int cur = kt & 1;
        const bool more = (kt + 1 < NT);
        if (more) {
            issueA(cur ^ 1, (kt + 1) * KTILE);  // DMA next A panel
            issueB((kt + 1) * KTILE);           // loads in flight over compute
        }
        compute(cur);
        if (more) storeB(cur ^ 1);
        wait_async0();
        __syncthreads();
    }

    // Epilogue: bias + clamp + GLU + router-weight scale, store bf16.
    #pragma unroll
    for (int sub = 0; sub < 2; ++sub) {
        const int col = i0 + cBase + sub * 16 + m;
        const float gb = gub[(size_t)e * (2 * DI) + 2 * col];
        const float ub = gub[(size_t)e * (2 * DI) + 2 * col + 1];
        const v8f g = sub ? accG1 : accG0;
        const v8f u = sub ? accU1 : accU0;
        #pragma unroll
        for (int r = 0; r < 8; ++r) {
            const int row = n0 + rBase + r + hi * 8;
            float gv = fminf(g[r] + gb, 7.f);
            float uv = fminf(fmaxf(u[r] + ub, -7.f), 7.f);
            const float sig = 1.f / (1.f + __expf(-1.702f * gv));
            const float act = (uv + 1.f) * gv * sig;
            const float wt  = wmask[(size_t)row * DE + e];
            actw[((size_t)row * DE + e) * DI + col] = (bf16)(act * wt);
        }
    }
}

// ---------------------------------------------------------------------------
// Kernel 3: down GEMM + combine over all experts (K = E*I = 8192), double
// buffered; A (bf16 activations) via async DMA, B register-staged.
// ---------------------------------------------------------------------------
__global__ __launch_bounds__(256) void moe_down_k(
    const bf16*  __restrict__ actw,     // [N,E,I] (weight-scaled)
    const float* __restrict__ dwn,      // [E,I,H]
    const float* __restrict__ db,       // [E,H]
    const float* __restrict__ wmask,    // [N,E]
    float* __restrict__ out)            // [N,H]
{
    const int n0 = blockIdx.x * 64;
    const int h0 = blockIdx.y * 64;

    __shared__ bf16 As[2][TILE_ELEMS];
    __shared__ bf16 Bs[2][TILE_ELEMS];

    const int tid   = threadIdx.x;
    const int lane  = tid & 31;
    const int wave  = tid >> 5;
    const int rBase = (wave >> 1) * 16;
    const int cBase = (wave & 1) * 32;
    const int m     = lane & 15;
    const int hi    = (lane >> 4) & 1;

    v8f acc0 = {}, acc1 = {};
    float t0[8], t1[8];

    auto issueA = [&](int buf, int kt) {    // kt in [0, 128): e = kt>>4
        const int e  = kt >> 4;
        const int k0 = (kt & 15) * KTILE;
        const unsigned base = lds_off(&As[buf][0]);
        #pragma unroll
        for (int it = 0; it < 2; ++it) {
            const int idx = tid + it * 256;
            const int r = idx >> 3, q = idx & 7;
            async_copy_b128(base + (unsigned)(r * (TSTRIDE * 2) + q * 16),
                            actw + ((size_t)(n0 + r) * DE + e) * DI + k0 + q * 8);
        }
    };
    auto issueB = [&](int kt) {
        const int e  = kt >> 4;
        const int k0 = (kt & 15) * KTILE;
        #pragma unroll
        for (int it = 0; it < 8; ++it) {
            const int idx = tid + it * 256;
            const int kp = idx >> 6, cc = idx & 63;
            const size_t base = ((size_t)e * DI + (k0 + kp * 2)) * DH + h0 + cc;
            t0[it] = dwn[base];
            t1[it] = dwn[base + DH];
        }
    };
    auto storeB = [&](int buf) {
        unsigned* b = (unsigned*)&Bs[buf][0];
        #pragma unroll
        for (int it = 0; it < 8; ++it) {
            const int idx = tid + it * 256;
            const int kp = idx >> 6, cc = idx & 63;
            b[cc * (TSTRIDE / 2) + kp] = pack_bf16(t0[it], t1[it]);
        }
    };
    auto compute = [&](int buf) {
        #pragma unroll
        for (int ks = 0; ks < KTILE; ks += 32) {
            const v16bf a  = load_frag(&As[buf][0], rBase + m,      ks, lane);
            const v16bf b0 = load_frag(&Bs[buf][0], cBase + m,      ks, lane);
            const v16bf b1 = load_frag(&Bs[buf][0], cBase + 16 + m, ks, lane);
            acc0 = wmma_bf16(a, b0, acc0);
            acc1 = wmma_bf16(a, b1, acc1);
        }
    };

    issueA(0, 0);
    issueB(0);
    storeB(0);
    wait_async0();
    __syncthreads();

    const int NT = DE * (DI / KTILE);     // 128 panels
    for (int kt = 0; kt < NT; ++kt) {
        const int cur = kt & 1;
        const bool more = (kt + 1 < NT);
        if (more) {
            issueA(cur ^ 1, kt + 1);
            issueB(kt + 1);
        }
        compute(cur);
        if (more) storeB(cur ^ 1);
        wait_async0();
        __syncthreads();
    }

    // Epilogue: add sum_e w[n,e]*db[e,h] and store fp32.
    #pragma unroll
    for (int sub = 0; sub < 2; ++sub) {
        const int col = h0 + cBase + sub * 16 + m;
        float dbv[DE];
        #pragma unroll
        for (int e = 0; e < DE; ++e) dbv[e] = db[(size_t)e * DH + col];
        const v8f a = sub ? acc1 : acc0;
        #pragma unroll
        for (int r = 0; r < 8; ++r) {
            const int row = n0 + rBase + r + hi * 8;
            float bias = 0.f;
            #pragma unroll
            for (int e = 0; e < DE; ++e)
                bias = fmaf(wmask[(size_t)row * DE + e], dbv[e], bias);
            out[(size_t)row * DH + col] = a[r] + bias;
        }
    }
}

// ---------------------------------------------------------------------------
extern "C" void kernel_launch(void* const* d_in, const int* in_sizes, int n_in,
                              void* d_out, int out_size, void* d_ws, size_t ws_size,
                              hipStream_t stream) {
    const float* x    = (const float*)d_in[0];   // [1,1024,1024]
    const float* rw   = (const float*)d_in[1];   // [8,1024]
    const float* gup  = (const float*)d_in[2];   // [8,1024,2048]
    const float* gub  = (const float*)d_in[3];   // [8,2048]
    const float* dwn  = (const float*)d_in[4];   // [8,1024,1024]
    const float* db   = (const float*)d_in[5];   // [8,1024]

    float* outp    = (float*)d_out;                       // [N,H]
    float* scoresp = outp + (size_t)DN * DH;              // [N,E]

    float* ws_w  = (float*)d_ws;                          // [N,E] weights
    bf16*  ws_xh = (bf16*)((char*)d_ws + 64 * 1024);      // [N,H] bf16 (2 MB)
    bf16*  ws_act = ws_xh + (size_t)DN * DH;              // [N,E,I] bf16 (16 MB)

    // 0) Convert hidden states to bf16 once.
    moe_cvt_k<<<(DN * DH) / (256 * 8), 256, 0, stream>>>(x, (unsigned*)ws_xh);

    // 1) Router: softmax scores (output #2) + masked top-2 weights.
    moe_router_k<<<DN / 32, 256, 0, stream>>>(x, rw, scoresp, ws_w);

    // 2) gate_up GEMM + activation, weight-scaled bf16 activations.
    moe_gateup_k<<<dim3(DN / 64, DI / 64, DE), 256, 0, stream>>>(
        ws_xh, gup, gub, ws_w, ws_act);

    // 3) down GEMM over all experts + bias combine.
    moe_down_k<<<dim3(DN / 64, DH / 64), 256, 0, stream>>>(
        ws_act, dwn, db, ws_w, outp);
}
